// NonCommutativeMultiHeadAttention_51711406244364
// MI455X (gfx1250) — matrix operand order repaired and checked by reading, not compiled, not run on hardware
//
#include <hip/hip_runtime.h>
#include <hip/hip_bf16.h>

// MHA: B=4, S=2048, D=1024, H=16, dk=64.  ~137 GFLOP vs ~80MB HBM -> compute
// bound -> bf16 WMMA (v_wmma_f32_16x16x32_bf16) with f32 accum.
// Waves are register-blocked (32x64 C per wave in GEMMs, 32 queries per wave
// in attention) so operand traffic is ~768B per WMMA instead of 2KB.

typedef __bf16 v16bf __attribute__((ext_vector_type(16)));
typedef float  v8f   __attribute__((ext_vector_type(8)));
typedef unsigned int v4u __attribute__((ext_vector_type(4)));

#define WMMA_BF16(a, b, c) \
  __builtin_amdgcn_wmma_f32_16x16x32_bf16(false, (a), false, (b), (short)0, (c), false, false)

// A-operand (16x32 bf16, MxK): lane m=(lane&15); VGPR0-3: K = half*8+0..7,
// VGPR4-7: K = 16+half*8+0..7  -> two 16B loads.
__device__ __forceinline__ v16bf ldfragA(const __bf16* p, int half) {
  union { v4u u[2]; v16bf v; } t;
  t.u[0] = *(const v4u*)(p + half * 8);
  t.u[1] = *(const v4u*)(p + 16 + half * 8);
  return t.v;
}
// B-operand (32x16 bf16, KxN): lane n=(lane&15); VGPR0-3: K = half*8+0..7,
// VGPR4-7: K = 16+half*8+0..7 -> two 16B loads.
__device__ __forceinline__ v16bf ldfragB(const __bf16* p, int half) {
  union { v4u u[2]; v16bf v; } t;
  t.u[0] = *(const v4u*)(p + half * 8);
  t.u[1] = *(const v4u*)(p + 16 + half * 8);
  return t.v;
}

// ---------------- prep: fp32 -> bf16 convert / transpose ----------------
__global__ void k_cvt_bf16(const float* __restrict__ in, unsigned short* __restrict__ out, int n) {
  int i = blockIdx.x * blockDim.x + threadIdx.x;
  if (i < n) ((__bf16*)out)[i] = (__bf16)in[i];
}

// out[n][k] = (bf16) in[k][n], 1024x1024, tiled through LDS
__global__ void k_transpose_bf16(const float* __restrict__ in, unsigned short* __restrict__ out) {
  __shared__ float tile[32][33];
  int bx = blockIdx.x * 32, by = blockIdx.y * 32;
  int tx = threadIdx.x & 31, ty = threadIdx.x >> 5;  // 32x8
#pragma unroll
  for (int i = ty; i < 32; i += 8) tile[i][tx] = in[(by + i) * 1024 + bx + tx];
  __syncthreads();
#pragma unroll
  for (int i = ty; i < 32; i += 8)
    ((__bf16*)out)[(bx + i) * 1024 + by + tx] = (__bf16)tile[tx][i];
}

// ---------------- fused QKV projection + elementwise transforms ----------------
// One wave = 32x64 C block (2M x 4N WMMA tiles).  t in [0, 3*4096): mat = Q/K/V.
// N block of 64 == one head.
__global__ void k_proj(const unsigned short* __restrict__ xbf,
                       const unsigned short* __restrict__ WqT,
                       const unsigned short* __restrict__ WkT,
                       const unsigned short* __restrict__ WvT,
                       const float* __restrict__ alphaP,
                       unsigned short* __restrict__ Qh,
                       unsigned short* __restrict__ Kh,
                       unsigned short* __restrict__ Vt) {
  int wave = threadIdx.x >> 5, lane = threadIdx.x & 31;
  int lane16 = lane & 15, half = lane >> 4;
  int t = blockIdx.x * 8 + wave;            // [0, 12288)
  int mat = t >> 12, tt = t & 4095;         // 4096 tiles / matrix
  int tileM = tt >> 4, tileN = tt & 15;     // 256 M-blocks x 16 N-blocks

  const __bf16* A0 = (const __bf16*)xbf + (size_t)(tileM * 32 + lane16) * 1024;
  const unsigned short* Wsel = (mat == 0) ? WqT : (mat == 1) ? WkT : WvT;
  const __bf16* B0 = (const __bf16*)Wsel + (size_t)(tileN * 64 + lane16) * 1024;

  v8f z = {0.f,0.f,0.f,0.f,0.f,0.f,0.f,0.f};
  v8f acc[2][4];
#pragma unroll
  for (int i = 0; i < 2; ++i)
#pragma unroll
    for (int j = 0; j < 4; ++j) acc[i][j] = z;

  for (int kk = 0; kk < 1024; kk += 32) {
    v16bf a0 = ldfragA(A0 + kk, half);
    v16bf a1 = ldfragA(A0 + 16 * 1024 + kk, half);
#pragma unroll
    for (int j = 0; j < 4; ++j) {
      v16bf b = ldfragB(B0 + (size_t)j * 16 * 1024 + kk, half);
      acc[0][j] = WMMA_BF16(a0, b, acc[0][j]);
      acc[1][j] = WMMA_BF16(a1, b, acc[1][j]);
    }
  }

  float alpha = alphaP[0];
#pragma unroll
  for (int i = 0; i < 2; ++i)
#pragma unroll
    for (int j = 0; j < 4; ++j)
#pragma unroll
      for (int r = 0; r < 8; ++r) {
        float c = acc[i][j][r];
        if (mat == 0)      c = c + 0.1f * __sinf(c);
        else if (mat == 1) c = c + 0.1f * __cosf(c);
        else { float e = __expf(2.0f * c); c = c + alpha * 0.1f * (1.0f - 2.0f / (e + 1.0f)); }
        int sg = tileM * 32 + i * 16 + r + 8 * half;   // global row in [0,8192)
        int b_ = sg >> 11, s = sg & 2047;
        int bh = b_ * 16 + tileN;                       // head == tileN
        int dcol = j * 16 + lane16;
        if (mat == 2)
          ((__bf16*)Vt)[((size_t)(bh * 64 + dcol)) * 2048 + s] = (__bf16)c;   // [bh][d][s]
        else {
          unsigned short* dst = (mat == 0) ? Qh : Kh;
          ((__bf16*)dst)[((size_t)(bh * 2048 + s)) * 64 + dcol] = (__bf16)c;  // [bh][s][d]
        }
      }
}

// ---------------- flash attention (online softmax), one wave = 32 queries ----------------
__global__ void k_attn(const unsigned short* __restrict__ Qh,
                       const unsigned short* __restrict__ Kh,
                       const unsigned short* __restrict__ Vt,
                       unsigned short* __restrict__ AO) {
  __shared__ alignas(16) unsigned short lds[4][32 * 32];  // per-wave 32x32 bf16 P tile
  int wave = threadIdx.x >> 5, lane = threadIdx.x & 31;
  int lane16 = lane & 15, half = lane >> 4;
  int qt = blockIdx.x * 4 + wave;
  int bh = qt >> 6;                  // 64 q-blocks (of 32) per (b,h)
  int q0 = (qt & 63) << 5;

  const __bf16* Qrow0 = (const __bf16*)Qh + (size_t)(bh * 2048 + q0 + lane16) * 64;
  v16bf qa[2][2];
#pragma unroll
  for (int i = 0; i < 2; ++i) {
    qa[i][0] = ldfragA(Qrow0 + (size_t)i * 16 * 64, half);       // d = 0..31
    qa[i][1] = ldfragA(Qrow0 + (size_t)i * 16 * 64 + 32, half);  // d = 32..63
  }

  v8f z = {0.f,0.f,0.f,0.f,0.f,0.f,0.f,0.f};
  v8f o[2][4];
  float mx[2][8], sm[2][8];
#pragma unroll
  for (int i = 0; i < 2; ++i)
#pragma unroll
    for (int r = 0; r < 8; ++r) { mx[i][r] = -3e38f; sm[i][r] = 0.f; }
#pragma unroll
  for (int i = 0; i < 2; ++i)
#pragma unroll
    for (int tt = 0; tt < 4; ++tt) o[i][tt] = z;

  __bf16* P = (__bf16*)lds[wave];
  const __bf16* Kbase = (const __bf16*)Kh + (size_t)bh * 2048 * 64;
  const __bf16* Vbase = (const __bf16*)Vt + (size_t)bh * 64 * 2048;

  for (int kt = 0; kt < 2048; kt += 32) {
    // scores: 2 q-subtiles x 2 key-tiles, K-dim = 64 (2 WMMA steps each)
    const __bf16* Kr0 = Kbase + (size_t)(kt + lane16) * 64;
    v16bf kb00 = ldfragB(Kr0, half);
    v16bf kb01 = ldfragB(Kr0 + 32, half);
    v16bf kb10 = ldfragB(Kr0 + 16 * 64, half);
    v16bf kb11 = ldfragB(Kr0 + 16 * 64 + 32, half);
    v8f c[2][2];
#pragma unroll
    for (int i = 0; i < 2; ++i) {
      c[i][0] = WMMA_BF16(qa[i][0], kb00, z);
      c[i][0] = WMMA_BF16(qa[i][1], kb01, c[i][0]);
      c[i][1] = WMMA_BF16(qa[i][0], kb10, z);
      c[i][1] = WMMA_BF16(qa[i][1], kb11, c[i][1]);
    }

#pragma unroll
    for (int i = 0; i < 2; ++i)
#pragma unroll
      for (int r = 0; r < 8; ++r) {
        float s0 = c[i][0][r] * 0.125f, s1 = c[i][1][r] * 0.125f;  // 1/sqrt(64)
        float cmx = fmaxf(s0, s1);                                  // reduce over 16-lane N group
        cmx = fmaxf(cmx, __shfl_xor(cmx, 1, 32));
        cmx = fmaxf(cmx, __shfl_xor(cmx, 2, 32));
        cmx = fmaxf(cmx, __shfl_xor(cmx, 4, 32));
        cmx = fmaxf(cmx, __shfl_xor(cmx, 8, 32));
        float nm = fmaxf(mx[i][r], cmx);
        float fac = __expf(mx[i][r] - nm);
        float p0 = __expf(s0 - nm), p1 = __expf(s1 - nm);
        float ps = p0 + p1;
        ps += __shfl_xor(ps, 1, 32);
        ps += __shfl_xor(ps, 2, 32);
        ps += __shfl_xor(ps, 4, 32);
        ps += __shfl_xor(ps, 8, 32);
        sm[i][r] = sm[i][r] * fac + ps;
        mx[i][r] = nm;
        o[i][0][r] *= fac; o[i][1][r] *= fac; o[i][2][r] *= fac; o[i][3][r] *= fac;
        int m = i * 16 + r + 8 * half;                 // C layout -> LDS [m][k]
        P[m * 32 + lane16] = (__bf16)p0;
        P[m * 32 + 16 + lane16] = (__bf16)p1;
      }

    // re-gather P as A-fragments (2 x 16x32) from LDS, then P @ V
    const __bf16* Vb = Vbase + (size_t)lane16 * 2048 + kt;
    v16bf pa0 = ldfragA(P + lane16 * 32, half);
    v16bf pa1 = ldfragA(P + (16 + lane16) * 32, half);
#pragma unroll
    for (int tt = 0; tt < 4; ++tt) {
      v16bf vb = ldfragB(Vb + (size_t)tt * 16 * 2048, half);
      o[0][tt] = WMMA_BF16(pa0, vb, o[0][tt]);
      o[1][tt] = WMMA_BF16(pa1, vb, o[1][tt]);
    }
  }

  int b_ = bh >> 4, h = bh & 15;
#pragma unroll
  for (int i = 0; i < 2; ++i)
#pragma unroll
    for (int r = 0; r < 8; ++r) {
      float inv = 1.0f / sm[i][r];
      int row = b_ * 2048 + q0 + i * 16 + r + 8 * half;
      __bf16* dst = (__bf16*)AO + (size_t)row * 1024 + h * 64 + lane16;
      dst[0]  = (__bf16)(o[i][0][r] * inv);
      dst[16] = (__bf16)(o[i][1][r] * inv);
      dst[32] = (__bf16)(o[i][2][r] * inv);
      dst[48] = (__bf16)(o[i][3][r] * inv);
    }
}

// ---------------- output projection: f32 result, 32x64 C per wave ----------------
__global__ void k_out(const unsigned short* __restrict__ AO,
                      const unsigned short* __restrict__ WoT,
                      float* __restrict__ out) {
  int wave = threadIdx.x >> 5, lane = threadIdx.x & 31;
  int lane16 = lane & 15, half = lane >> 4;
  int t = blockIdx.x * 8 + wave;            // [0, 4096)
  int tileM = t >> 4, tileN = t & 15;
  const __bf16* A0 = (const __bf16*)AO  + (size_t)(tileM * 32 + lane16) * 1024;
  const __bf16* B0 = (const __bf16*)WoT + (size_t)(tileN * 64 + lane16) * 1024;

  v8f z = {0.f,0.f,0.f,0.f,0.f,0.f,0.f,0.f};
  v8f acc[2][4];
#pragma unroll
  for (int i = 0; i < 2; ++i)
#pragma unroll
    for (int j = 0; j < 4; ++j) acc[i][j] = z;

  for (int kk = 0; kk < 1024; kk += 32) {
    v16bf a0 = ldfragA(A0 + kk, half);
    v16bf a1 = ldfragA(A0 + 16 * 1024 + kk, half);
#pragma unroll
    for (int j = 0; j < 4; ++j) {
      v16bf b = ldfragB(B0 + (size_t)j * 16 * 1024 + kk, half);
      acc[0][j] = WMMA_BF16(a0, b, acc[0][j]);
      acc[1][j] = WMMA_BF16(a1, b, acc[1][j]);
    }
  }
#pragma unroll
  for (int i = 0; i < 2; ++i)
#pragma unroll
    for (int j = 0; j < 4; ++j)
#pragma unroll
      for (int r = 0; r < 8; ++r)
        out[(size_t)(tileM * 32 + i * 16 + r + 8 * half) * 1024 + tileN * 64 + j * 16 + lane16] =
            acc[i][j][r];
}

extern "C" void kernel_launch(void* const* d_in, const int* in_sizes, int n_in,
                              void* d_out, int out_size, void* d_ws, size_t ws_size,
                              hipStream_t stream) {
  const float* x  = (const float*)d_in[0];
  const float* Wq = (const float*)d_in[1];
  const float* Wk = (const float*)d_in[2];
  const float* Wv = (const float*)d_in[3];
  const float* Wo = (const float*)d_in[4];
  const float* alpha = (const float*)d_in[5];
  float* out = (float*)d_out;

  char* ws = (char*)d_ws;
  const size_t MB = 1u << 20;
  unsigned short* xbf = (unsigned short*)(ws + 0);        // 16 MB
  unsigned short* WqT = (unsigned short*)(ws + 16 * MB);  //  2 MB each
  unsigned short* WkT = (unsigned short*)(ws + 18 * MB);
  unsigned short* WvT = (unsigned short*)(ws + 20 * MB);
  unsigned short* WoT = (unsigned short*)(ws + 22 * MB);
  unsigned short* Qh  = (unsigned short*)(ws + 24 * MB);  // 16 MB [bh][s][64]
  unsigned short* Kh  = (unsigned short*)(ws + 40 * MB);  // 16 MB [bh][s][64]
  unsigned short* Vt  = (unsigned short*)(ws + 56 * MB);  // 16 MB [bh][64][s]
  unsigned short* AO  = (unsigned short*)(ws + 72 * MB);  // 16 MB [row][1024]

  const int NX = 4 * 2048 * 1024;
  k_cvt_bf16<<<NX / 256, 256, 0, stream>>>(x, xbf, NX);
  dim3 tg(32, 32);
  k_transpose_bf16<<<tg, 256, 0, stream>>>(Wq, WqT);
  k_transpose_bf16<<<tg, 256, 0, stream>>>(Wk, WkT);
  k_transpose_bf16<<<tg, 256, 0, stream>>>(Wv, WvT);
  k_transpose_bf16<<<tg, 256, 0, stream>>>(Wo, WoT);

  k_proj<<<1536, 256, 0, stream>>>(xbf, WqT, WkT, WvT, alpha, Qh, Kh, Vt); // 3*4096 wave-tiles / 8
  k_attn<<<1024, 128, 0, stream>>>(Qh, Kh, Vt, AO);                         // 4096 wave-tiles / 4
  k_out<<<512, 256, 0, stream>>>(AO, WoT, out);                             // 4096 wave-tiles / 8
}
